// DMPNNEncoder_3066606649635
// MI455X (gfx1250) — compile-verified, matrix operand order checked by reading, not run on hardware
//
#include <hip/hip_runtime.h>
#include <hip/hip_bf16.h>

// ---------------------------------------------------------------------------
// DMPNN encoder for gfx1250 (MI455X). bf16 WMMA (v_wmma_f32_16x16x32_bf16)
// for all GEMMs; tiles staged with CDNA5 async global->LDS DMA
// (global_load_async_to_lds_b128, ASYNCcnt) so no VGPR staging / no spills.
// ---------------------------------------------------------------------------

typedef __attribute__((ext_vector_type(16))) __bf16 bf16x16;
typedef __attribute__((ext_vector_type(8)))  float  f32x8;

#define N_ATOMS   120000
#define N_BONDS   240000
#define HIDDEN    256
#define ATOM_DIM  133
#define BOND_DIM  14
#define MOL_DIM   200
#define N_MOL     4096
#define MAX_DEG   6
#define OUTC      (HIDDEN + MOL_DIM)   // 456
#define K_INIT    160                  // pad(147) -> %32==0
#define K_MSG     256
#define K_ATOM    416                  // pad(389) -> %32==0
#define NEG_SLOPE 0.01f
#define EPS       1e-5f

// ------------------------------ GEMM -------------------------------------
// Y[M][256] = leaky( A[M][K]*Wb[256][K]^T + bias (+ addend) )
// Tile 128(M) x 128(N); 8 waves in 4x2; each wave 32x64 = 2x4 WMMA tiles.
#define MT 128
#define NT 128
#define KT 32
#define LDS_STRIDE 40   // 32 + 8 pad halves, kills LDS bank conflicts

union FragU { uint4 u[2]; bf16x16 v; };

// CDNA5 async DMA: global -> LDS, 16B per lane, tracked by ASYNCcnt.
__device__ __forceinline__ void async_ld16(void* lds_ptr, const void* gptr) {
    unsigned lds_addr = (unsigned)(uintptr_t)lds_ptr;        // flat LDS: offset in addr[31:0]
    unsigned long long ga = (unsigned long long)(uintptr_t)gptr;
    asm volatile("global_load_async_to_lds_b128 %0, %1, off"
                 :: "v"(lds_addr), "v"(ga) : "memory");
}
__device__ __forceinline__ void wait_async0() {
    asm volatile("s_wait_asynccnt 0x0" ::: "memory");
}

template<bool HAS_ADD>
__global__ __launch_bounds__(256)
void gemm_bf16_leaky(const __bf16* __restrict__ A, int lda,
                     const __bf16* __restrict__ W, int ldw,
                     const float* __restrict__ bias,
                     const float* __restrict__ addend,
                     float* __restrict__ Y,
                     int M, int K)
{
    constexpr int N = HIDDEN;
    __shared__ __bf16 As[2][MT][LDS_STRIDE];
    __shared__ __bf16 Bs[2][NT][LDS_STRIDE];

    const int tid  = threadIdx.x;
    const int lane = tid & 31;
    const int wave = tid >> 5;
    const int wm   = wave & 3;     // 0..3 -> 32-row band
    const int wn   = wave >> 2;    // 0..1 -> 64-col band
    const int bm   = blockIdx.x * MT;
    const int bnn  = blockIdx.y * NT;

    // staging indices: 128 rows x 4 uint4/row = 512 vectors, 2 per thread
    const int r0 = tid >> 2;             // 0..63
    const int c0 = (tid & 3) * 8;        // half-offset within row
    const int r1 = r0 + 64;              // 64..127
    int ga0 = bm + r0; if (ga0 > M - 1) ga0 = M - 1;   // branch-free clamp
    int ga1 = bm + r1; if (ga1 > M - 1) ga1 = M - 1;
    const __bf16* arow0 = A + (size_t)ga0 * lda + c0;
    const __bf16* arow1 = A + (size_t)ga1 * lda + c0;
    const __bf16* brow0 = W + (size_t)(bnn + r0) * ldw + c0;
    const __bf16* brow1 = W + (size_t)(bnn + r1) * ldw + c0;

    f32x8 acc[2][4] = {};

    const int mrow  = lane & 15;
    const int khalf = (lane >> 4) * 8;   // ISA A/B frag: K sub-half base
    const int nk = K / KT;

    // ---- prologue: DMA tile 0 into buffer 0 ----
    async_ld16(&As[0][r0][c0], arow0);
    async_ld16(&As[0][r1][c0], arow1);
    async_ld16(&Bs[0][r0][c0], brow0);
    async_ld16(&Bs[0][r1][c0], brow1);

    for (int k = 0; k < nk; ++k) {
        const int p = k & 1;
        wait_async0();       // tile k has landed in LDS buffer p
        __syncthreads();     // publish buffer p; readers of p^1 are done

        // ---- kick off DMA for tile k+1 into the other buffer ----
        if (k + 1 < nk) {
            const int k0 = (k + 1) * KT;
            const int q = p ^ 1;
            async_ld16(&As[q][r0][c0], arow0 + k0);
            async_ld16(&As[q][r1][c0], arow1 + k0);
            async_ld16(&Bs[q][r0][c0], brow0 + k0);
            async_ld16(&Bs[q][r1][c0], brow1 + k0);
        }

        // ---- fragments (ISA 16-bit layout: lanes 0-15 K=0..7 & 16..23;
        //      lanes 16-31 K=8..15 & 24..31; B symmetric, N as rows) ----
        FragU a[2];
        #pragma unroll
        for (int i = 0; i < 2; ++i) {
            const __bf16* ap = &As[p][wm * 32 + i * 16 + mrow][0];
            a[i].u[0] = *(const uint4*)(ap + khalf);
            a[i].u[1] = *(const uint4*)(ap + khalf + 16);
        }
        #pragma unroll
        for (int j = 0; j < 4; ++j) {
            FragU b;
            const __bf16* bp = &Bs[p][wn * 64 + j * 16 + mrow][0];
            b.u[0] = *(const uint4*)(bp + khalf);
            b.u[1] = *(const uint4*)(bp + khalf + 16);
            #pragma unroll
            for (int i = 0; i < 2; ++i)
                acc[i][j] = __builtin_amdgcn_wmma_f32_16x16x32_bf16(
                    false, a[i].v, false, b.v,
                    (short)0, acc[i][j], false, false);
        }
    }

    // ---- epilogue: +bias (+addend) -> leaky -> f32 store ----
    const int rowsub = (lane >> 4) * 8;    // C layout: lanes 16-31 hold M+8
    #pragma unroll
    for (int i = 0; i < 2; ++i) {
        #pragma unroll
        for (int j = 0; j < 4; ++j) {
            int col = bnn + wn * 64 + j * 16 + (lane & 15);
            float bv = bias[col];
            #pragma unroll
            for (int r = 0; r < 8; ++r) {
                int row = bm + wm * 32 + i * 16 + rowsub + r;
                if (row < M) {
                    float z = acc[i][j][r] + bv;
                    if (HAS_ADD) z += addend[(size_t)row * N + col];
                    Y[(size_t)row * N + col] = (z >= 0.0f) ? z : NEG_SLOPE * z;
                }
            }
        }
    }
}

// ------------------------- BatchNorm stats/apply ---------------------------
__global__ void zero_f32(float* p, int n) {
    int i = blockIdx.x * blockDim.x + threadIdx.x;
    if (i < n) p[i] = 0.0f;
}

__global__ __launch_bounds__(HIDDEN)
void col_stats(const float* __restrict__ Y, int M, float* __restrict__ stats) {
    int col = threadIdx.x;
    float s = 0.0f, s2 = 0.0f;
    for (int row = blockIdx.x; row < M; row += gridDim.x) {
        float v = Y[(size_t)row * HIDDEN + col];
        s += v; s2 += v * v;
    }
    atomicAdd(&stats[col], s);
    atomicAdd(&stats[HIDDEN + col], s2);
}

__global__ __launch_bounds__(HIDDEN)
void bn_apply(const float* __restrict__ Y, const float* __restrict__ stats,
              const float* __restrict__ w, const float* __restrict__ b,
              float* __restrict__ H, float* __restrict__ H2, int M)
{
    int col = threadIdx.x;
    float inv_m = 1.0f / (float)M;
    float mu  = stats[col] * inv_m;
    float var = stats[HIDDEN + col] * inv_m - mu * mu;
    float scale = w[col] * rsqrtf(var + EPS);
    float shift = b[col];
    for (int row = blockIdx.x; row < M; row += gridDim.x) {
        size_t idx = (size_t)row * HIDDEN + col;
        float o = (Y[idx] - mu) * scale + shift;
        H[idx] = o;
        if (H2) H2[idx] = o;
    }
}

// --------------------------- gather / concat -------------------------------
__global__ __launch_bounds__(HIDDEN)
void atom_sum(const float* __restrict__ h, const int* __restrict__ aibm,
              float* __restrict__ out)
{
    int a = blockIdx.x, c = threadIdx.x;
    float s = 0.0f;
    #pragma unroll
    for (int d = 0; d < MAX_DEG; ++d) {
        int bid = aibm[a * MAX_DEG + d];     // 1-based, 0 = pad
        if (bid > 0) s += h[(size_t)(bid - 1) * HIDDEN + c];
    }
    out[(size_t)a * HIDDEN + c] = s;
}

__global__ __launch_bounds__(HIDDEN)
void make_msg(const float* __restrict__ asum, const float* __restrict__ h,
              const int* __restrict__ src, const int* __restrict__ rev,
              __bf16* __restrict__ m)
{
    int bnd = blockIdx.x, c = threadIdx.x;
    float v = asum[(size_t)src[bnd] * HIDDEN + c]
            - h[(size_t)rev[bnd] * HIDDEN + c];
    m[(size_t)bnd * HIDDEN + c] = (__bf16)v;
}

__global__ __launch_bounds__(K_INIT)
void build_x0(const float* __restrict__ bondf, const float* __restrict__ atomf,
              const int* __restrict__ src, __bf16* __restrict__ x0)
{
    int bnd = blockIdx.x, t = threadIdx.x;     // 160 threads
    float v = 0.0f;
    if (t < BOND_DIM)                   v = bondf[(size_t)bnd * BOND_DIM + t];
    else if (t < BOND_DIM + ATOM_DIM)   v = atomf[(size_t)src[bnd] * ATOM_DIM + (t - BOND_DIM)];
    x0[(size_t)bnd * K_INIT + t] = (__bf16)v;
}

__global__ __launch_bounds__(K_ATOM)
void build_xa(const float* __restrict__ atomf, const float* __restrict__ mv,
              __bf16* __restrict__ xa)
{
    int a = blockIdx.x, t = threadIdx.x;       // 416 threads
    float v = 0.0f;
    if (t < ATOM_DIM)                   v = atomf[(size_t)a * ATOM_DIM + t];
    else if (t < ATOM_DIM + HIDDEN)     v = mv[(size_t)a * HIDDEN + (t - ATOM_DIM)];
    xa[(size_t)a * K_ATOM + t] = (__bf16)v;
}

__global__ void cvt_weight(const float* __restrict__ W, __bf16* __restrict__ Wb,
                           int rows, int ksrc, int kdst)
{
    int i = blockIdx.x * blockDim.x + threadIdx.x;
    if (i >= rows * kdst) return;
    int r = i / kdst, k = i - r * kdst;
    Wb[i] = (__bf16)((k < ksrc) ? W[(size_t)r * ksrc + k] : 0.0f);
}

// --------------------------- output assembly -------------------------------
__global__ __launch_bounds__(512)
void init_out(float* __restrict__ out, const float* __restrict__ molf)
{
    int m = blockIdx.x, c = threadIdx.x;
    if (c < HIDDEN)       out[(size_t)m * OUTC + c] = 0.0f;
    else if (c < OUTC)    out[(size_t)m * OUTC + c] = molf[(size_t)m * MOL_DIM + (c - HIDDEN)];
}

__global__ __launch_bounds__(HIDDEN)
void pool_mol(const float* __restrict__ hv, const int* __restrict__ a2m,
              float* __restrict__ out)
{
    int a = blockIdx.x, c = threadIdx.x;
    atomicAdd(&out[(size_t)a2m[a] * OUTC + c], hv[(size_t)a * HIDDEN + c]);
}

// ------------------------------ launcher -----------------------------------
extern "C" void kernel_launch(void* const* d_in, const int* in_sizes, int n_in,
                              void* d_out, int out_size, void* d_ws, size_t ws_size,
                              hipStream_t stream)
{
    const float* atomf = (const float*)d_in[0];
    const float* bondf = (const float*)d_in[1];
    const int*   bidx  = (const int*)  d_in[2];   // [2][N_BONDS]; row0 = src
    const float* molf  = (const float*)d_in[3];
    const int*   aibm  = (const int*)  d_in[4];
    const int*   rev   = (const int*)  d_in[5];
    const int*   a2m   = (const int*)  d_in[6];
    const float* Wi    = (const float*)d_in[7];
    const float* Wib   = (const float*)d_in[8];
    const float* Wm    = (const float*)d_in[9];
    const float* Wmb   = (const float*)d_in[10];
    const float* Wa    = (const float*)d_in[11];
    const float* Wab   = (const float*)d_in[12];
    const float* bnw   = (const float*)d_in[13];
    const float* bnb   = (const float*)d_in[14];
    const int* src = bidx;                         // first N_BONDS ints

    // ---- workspace carve-up (all offsets 256B-aligned) ----
    char* ws = (char*)d_ws;
    size_t o_x  = 0;                                         // shared x0 / xa buf
    size_t sz_x = (size_t)N_ATOMS * K_ATOM * 2;              // 99,840,000 (>= x0 size)
    size_t o_y  = o_x + sz_x;
    size_t sz_h = (size_t)N_BONDS * HIDDEN * 4;              // 245,760,000
    size_t o_h0 = o_y + sz_h;
    size_t o_h  = o_h0 + sz_h;
    size_t o_as = o_h + sz_h;                                // atom sums / h_v
    size_t sz_a = (size_t)N_ATOMS * HIDDEN * 4;              // 122,880,000
    size_t o_m  = o_as + sz_a;                               // bf16 messages
    size_t o_wi = o_m + (size_t)N_BONDS * HIDDEN * 2;
    size_t o_wm = o_wi + (size_t)HIDDEN * K_INIT * 2;
    size_t o_wa = o_wm + (size_t)HIDDEN * K_MSG * 2;
    size_t o_st = o_wa + (size_t)HIDDEN * K_ATOM * 2;

    __bf16* xbuf  = (__bf16*)(ws + o_x);
    float*  ybuf  = (float*) (ws + o_y);
    float*  h0    = (float*) (ws + o_h0);
    float*  h     = (float*) (ws + o_h);
    float*  asum  = (float*) (ws + o_as);
    __bf16* mbuf  = (__bf16*)(ws + o_m);
    __bf16* wi_b  = (__bf16*)(ws + o_wi);
    __bf16* wm_b  = (__bf16*)(ws + o_wm);
    __bf16* wa_b  = (__bf16*)(ws + o_wa);
    float*  stats = (float*) (ws + o_st);

    // ---- weight conversion (bf16, K padded) ----
    cvt_weight<<<(HIDDEN * K_INIT + 255) / 256, 256, 0, stream>>>(Wi, wi_b, HIDDEN, ATOM_DIM + BOND_DIM, K_INIT);
    cvt_weight<<<(HIDDEN * K_MSG  + 255) / 256, 256, 0, stream>>>(Wm, wm_b, HIDDEN, HIDDEN, K_MSG);
    cvt_weight<<<(HIDDEN * K_ATOM + 255) / 256, 256, 0, stream>>>(Wa, wa_b, HIDDEN, ATOM_DIM + HIDDEN, K_ATOM);

    dim3 gemm_grid_b((N_BONDS + MT - 1) / MT, HIDDEN / NT);
    dim3 gemm_grid_a((N_ATOMS + MT - 1) / MT, HIDDEN / NT);

    // ---- stage 1: h0 = BN(leaky([bond | atom[src]] @ Wi^T + b)) ----
    build_x0<<<N_BONDS, K_INIT, 0, stream>>>(bondf, atomf, src, xbuf);
    zero_f32<<<2, 256, 0, stream>>>(stats, 2 * HIDDEN);
    gemm_bf16_leaky<false><<<gemm_grid_b, 256, 0, stream>>>(xbuf, K_INIT, wi_b, K_INIT,
                                                            Wib, nullptr, ybuf,
                                                            N_BONDS, K_INIT);
    col_stats<<<1024, HIDDEN, 0, stream>>>(ybuf, N_BONDS, stats);
    bn_apply<<<1024, HIDDEN, 0, stream>>>(ybuf, stats, bnw, bnb, h0, h, N_BONDS);

    // ---- message passing loop ----
    for (int d = 0; d < 3; ++d) {
        atom_sum<<<N_ATOMS, HIDDEN, 0, stream>>>(h, aibm, asum);
        make_msg<<<N_BONDS, HIDDEN, 0, stream>>>(asum, h, src, rev, mbuf);
        zero_f32<<<2, 256, 0, stream>>>(stats, 2 * HIDDEN);
        gemm_bf16_leaky<true><<<gemm_grid_b, 256, 0, stream>>>(mbuf, K_MSG, wm_b, K_MSG,
                                                               Wmb, h0, ybuf,
                                                               N_BONDS, K_MSG);
        col_stats<<<1024, HIDDEN, 0, stream>>>(ybuf, N_BONDS, stats);
        bn_apply<<<1024, HIDDEN, 0, stream>>>(ybuf, stats, bnw, bnb, h, nullptr, N_BONDS);
    }

    // ---- readout: h_v = BN(leaky([atom | m_v] @ Wa^T + b)) ----
    atom_sum<<<N_ATOMS, HIDDEN, 0, stream>>>(h, aibm, asum);
    build_xa<<<N_ATOMS, K_ATOM, 0, stream>>>(atomf, asum, xbuf);
    zero_f32<<<2, 256, 0, stream>>>(stats, 2 * HIDDEN);
    gemm_bf16_leaky<false><<<gemm_grid_a, 256, 0, stream>>>(xbuf, K_ATOM, wa_b, K_ATOM,
                                                            Wab, nullptr, ybuf,
                                                            N_ATOMS, K_ATOM);
    col_stats<<<1024, HIDDEN, 0, stream>>>(ybuf, N_ATOMS, stats);
    bn_apply<<<1024, HIDDEN, 0, stream>>>(ybuf, stats, bnw, bnb, asum, nullptr, N_ATOMS);

    // ---- molecule pooling + feature concat ----
    init_out<<<N_MOL, 512, 0, stream>>>((float*)d_out, molf);
    pool_mol<<<N_ATOMS, HIDDEN, 0, stream>>>(asum, a2m, (float*)d_out);
}
